// MultiheadAttention_8323646620266
// MI455X (gfx1250) — matrix operand order repaired and checked by reading, not compiled, not run on hardware
//
#include <hip/hip_runtime.h>
#include <stdint.h>

// ---- problem constants (match reference) ----
#define SH 8     // heads
#define SD 512   // feature dim
#define SK 64    // key feature dim
#define SF 64    // ffn dim
#define SL 512   // sequence length
#define SB 32    // batch

typedef __attribute__((ext_vector_type(16))) __bf16 v16bf;
typedef __attribute__((ext_vector_type(2)))  __bf16 v2bf;
typedef __attribute__((ext_vector_type(8)))  float  v8f;

// ---------------- scalar helpers (native bf16 converts) ----------------
__device__ __forceinline__ unsigned short f2bf(float f) {
  union { __bf16 h; unsigned short u; } c;
  c.h = (__bf16)f;
  return c.u;
}
__device__ __forceinline__ float bf2f(unsigned short u) {
  union { unsigned short u; __bf16 h; } c;
  c.u = u;
  return (float)c.h;
}
#if __has_builtin(__builtin_amdgcn_cvt_pk_bf16_f32)
__device__ __forceinline__ unsigned cvt2(float a, float b) {
  union { v2bf v; unsigned u; } c;
  c.v = __builtin_amdgcn_cvt_pk_bf16_f32(a, b);
  return c.u;
}
#else
// builtin not declared on this toolchain: emit the gfx1250 packed convert directly
__device__ __forceinline__ unsigned cvt2(float a, float b) {
  unsigned r;
  asm("v_cvt_pk_bf16_f32 %0, %1, %2" : "=v"(r) : "v"(a), "v"(b));
  return r;
}
#endif
__device__ __forceinline__ v8f vzero() {
  v8f z = {0.f,0.f,0.f,0.f,0.f,0.f,0.f,0.f};
  return z;
}
__device__ __forceinline__ uint4 pack8(float a0,float a1,float a2,float a3,
                                       float a4,float a5,float a6,float a7) {
  uint4 o;
  o.x = cvt2(a0, a1);
  o.y = cvt2(a2, a3);
  o.z = cvt2(a4, a5);
  o.w = cvt2(a6, a7);
  return o;
}

// ---------------- WMMA frag loaders (ISA 7.12.2 layouts, wave32) ----------------
// A 16x32 bf16: lanes 0-15 hold row M=lane, K = {kb..kb+7, kb+16..kb+23}, kb = 8*(lane>=16)
__device__ __forceinline__ v16bf load_a_frag(const unsigned short* a, int strideH,
                                             int m0, int kk, int lane) {
  const unsigned short* p = a + (long)(m0 + (lane & 15)) * strideH + kk + ((lane >> 4) << 3);
  union { v16bf v; uint4 q[2]; } u;
  u.q[0] = *(const uint4*)(p);
  u.q[1] = *(const uint4*)(p + 16);
  return u.v;
}
// B 32x16 bf16 in LDS as Btile[n][k]: lane holds col N=lane&15, K = {kb..kb+7, kb+16..kb+23}, kb = 8*(lane>=16)
__device__ __forceinline__ v16bf load_b_frag(const unsigned short* b, int strideH,
                                             int n0, int lane) {
  const unsigned short* p = b + (long)(n0 + (lane & 15)) * strideH + ((lane >> 4) << 3);
  union { v16bf v; uint4 q[2]; } u;
  u.q[0] = *(const uint4*)(p);
  u.q[1] = *(const uint4*)(p + 16);
  return u.v;
}
__device__ __forceinline__ v8f wmma_bf16(v16bf a, v16bf b, v8f c) {
  return __builtin_amdgcn_wmma_f32_16x16x32_bf16(false, a, false, b, (short)0, c, false, false);
}

// ---------------- cooperative tile loaders (256 threads) ----------------
// 64x32 f32 global -> bf16 LDS (stride 40 halves)
__device__ __forceinline__ void ld_f32_tile64(unsigned short* dst, const float* src,
                                              long ldS, int tid) {
  int row = tid >> 2, seg = (tid & 3) * 8;
  const float* g = src + (long)row * ldS + seg;
  float4 f0 = *(const float4*)(g);
  float4 f1 = *(const float4*)(g + 4);
  *(uint4*)(dst + row * 40 + seg) = pack8(f0.x,f0.y,f0.z,f0.w,f1.x,f1.y,f1.z,f1.w);
}
// 64x32 f32 global, per-column scale -> bf16 LDS
__device__ __forceinline__ void ld_f32_tile64_scaled(unsigned short* dst, const float* src,
                                                     long ldS, const float* scl, int tid) {
  int row = tid >> 2, seg = (tid & 3) * 8;
  const float* g = src + (long)row * ldS + seg;
  float4 f0 = *(const float4*)(g);
  float4 f1 = *(const float4*)(g + 4);
  *(uint4*)(dst + row * 40 + seg) =
      pack8(f0.x*scl[seg+0], f0.y*scl[seg+1], f0.z*scl[seg+2], f0.w*scl[seg+3],
            f1.x*scl[seg+4], f1.y*scl[seg+5], f1.z*scl[seg+6], f1.w*scl[seg+7]);
}
// 64x32 bf16 global (row-major, ldS in halves) -> LDS stride 40
__device__ __forceinline__ void ld_bf16_tile64(unsigned short* dst, const unsigned short* src,
                                               long ldS, int tid) {
  int row = tid >> 2, seg = (tid & 3) * 8;
  uint4 v = *(const uint4*)(src + (long)row * ldS + seg);
  *(uint4*)(dst + row * 40 + seg) = v;
}
// 32x32 bf16 global -> LDS stride 40
__device__ __forceinline__ void ld_bf16_tile32(unsigned short* dst, const unsigned short* src,
                                               long ldS, int tid) {
  int row = tid >> 3, seg = (tid & 7) * 4;
  uint2 v = *(const uint2*)(src + (long)row * ldS + seg);
  *(uint2*)(dst + row * 40 + seg) = v;
}
// 32(K) x 64(N) bf16 global tile -> transposed LDS Btile[n][k] stride 40
__device__ __forceinline__ void ld_bf16_tileT(unsigned short* dst, const unsigned short* src,
                                              long ldS, int tid) {
  int k = tid >> 3, seg = (tid & 7) * 8;
  uint4 v = *(const uint4*)(src + (long)k * ldS + seg);
  const unsigned short* h = (const unsigned short*)&v;
#pragma unroll
  for (int j = 0; j < 8; ++j) dst[(seg + j) * 40 + k] = h[j];
}

// ==================== kernel 0: zero ====================
__global__ __launch_bounds__(256) void k_zero(float* p, int n) {
  int i = blockIdx.x * 256 + threadIdx.x;
  if (i < n) p[i] = 0.f;
}

// ==================== kernel 1: Q/K projection + row l2-norm ====================
// Out[h,b,l,k] = l2norm_row( X[l,b,:] . W[h,k,:] ), bf16.  grid (L/64, B, H)
__global__ __launch_bounds__(256) void k_proj_norm(const float* __restrict__ X,
                                                   const float* __restrict__ W,
                                                   unsigned short* __restrict__ Out) {
  __shared__ __align__(16) unsigned short At[2][64 * 40];
  __shared__ __align__(16) unsigned short Bt[2][64 * 40];
  __shared__ float Ct[64 * 68];
  __shared__ float Red[64 * 4];
  __shared__ float Scale[64];
  int tid = threadIdx.x, lane = tid & 31, w = tid >> 5;
  int l0 = blockIdx.x * 64, b = blockIdx.y, h = blockIdx.z;
  const float* Ab = X + ((long)l0 * SB + b) * SD;
  const float* Wb = W + (long)h * SK * SD;
  int msub = (w & 3) * 16, n0 = (w >> 2) * 32;
  v8f acc0 = vzero(), acc1 = vzero();
  ld_f32_tile64(At[0], Ab, (long)SB * SD, tid);
  ld_f32_tile64(Bt[0], Wb, SD, tid);
  __syncthreads();
  const int NIT = SD / 32;
  for (int i = 0; i < NIT; ++i) {
    int cur = i & 1;
    if (i + 1 < NIT) {
      ld_f32_tile64(At[cur ^ 1], Ab + (i + 1) * 32, (long)SB * SD, tid);
      ld_f32_tile64(Bt[cur ^ 1], Wb + (i + 1) * 32, SD, tid);
    }
    v16bf a  = load_a_frag(At[cur], 40, msub, 0, lane);
    v16bf b0 = load_b_frag(Bt[cur], 40, n0, lane);
    v16bf b1 = load_b_frag(Bt[cur], 40, n0 + 16, lane);
    acc0 = wmma_bf16(a, b0, acc0);
    acc1 = wmma_bf16(a, b1, acc1);
    __syncthreads();
  }
  int mb = msub + ((lane >> 4) << 3), nc = lane & 15;
#pragma unroll
  for (int r = 0; r < 8; ++r) {
    Ct[(mb + r) * 68 + n0 + nc]      = acc0[r];
    Ct[(mb + r) * 68 + n0 + 16 + nc] = acc1[r];
  }
  __syncthreads();
  int rr = tid >> 2, q = tid & 3;
  float s = 0.f;
#pragma unroll
  for (int c = 0; c < 16; ++c) { float v = Ct[rr * 68 + q * 16 + c]; s += v * v; }
  Red[rr * 4 + q] = s;
  __syncthreads();
  if (q == 0) {
    float t = Red[rr * 4] + Red[rr * 4 + 1] + Red[rr * 4 + 2] + Red[rr * 4 + 3];
    Scale[rr] = 1.0f / fmaxf(sqrtf(t), 1e-12f);
  }
  __syncthreads();
  unsigned short* ob = Out + (((long)h * SB + b) * SL + l0) * SK;
#pragma unroll
  for (int c = 0; c < 16; ++c) {
    int col = q * 16 + c;
    ob[(long)rr * SK + col] = f2bf(Ct[rr * 68 + col] * Scale[rr]);
  }
}

// ==================== kernel 2: V projection ====================
// wv[h,b,l,e] = value[l,b,:] . WV[h,e,:], bf16.  grid (L/64, D/64, H*B)
__global__ __launch_bounds__(256) void k_proj_v(const float* __restrict__ V,
                                                const float* __restrict__ WV,
                                                unsigned short* __restrict__ wv) {
  __shared__ __align__(16) unsigned short At[2][64 * 40];
  __shared__ __align__(16) unsigned short Bt[2][64 * 40];
  int tid = threadIdx.x, lane = tid & 31, w = tid >> 5;
  int l0 = blockIdx.x * 64, n0t = blockIdx.y * 64;
  int z = blockIdx.z, b = z & (SB - 1), h = z >> 5;
  const float* Ab = V + ((long)l0 * SB + b) * SD;
  const float* Wb = WV + (long)h * SD * SD + (long)n0t * SD;
  int msub = (w & 3) * 16, n0 = (w >> 2) * 32;
  v8f acc0 = vzero(), acc1 = vzero();
  ld_f32_tile64(At[0], Ab, (long)SB * SD, tid);
  ld_f32_tile64(Bt[0], Wb, SD, tid);
  __syncthreads();
  const int NIT = SD / 32;
  for (int i = 0; i < NIT; ++i) {
    int cur = i & 1;
    if (i + 1 < NIT) {
      ld_f32_tile64(At[cur ^ 1], Ab + (i + 1) * 32, (long)SB * SD, tid);
      ld_f32_tile64(Bt[cur ^ 1], Wb + (i + 1) * 32, SD, tid);
    }
    v16bf a  = load_a_frag(At[cur], 40, msub, 0, lane);
    v16bf b0 = load_b_frag(Bt[cur], 40, n0, lane);
    v16bf b1 = load_b_frag(Bt[cur], 40, n0 + 16, lane);
    acc0 = wmma_bf16(a, b0, acc0);
    acc1 = wmma_bf16(a, b1, acc1);
    __syncthreads();
  }
  unsigned short* dst = wv + ((long)z * SL + l0) * SD + n0t;
  int mb = msub + ((lane >> 4) << 3), nc = lane & 15;
#pragma unroll
  for (int r = 0; r < 8; ++r) {
    dst[(long)(mb + r) * SD + n0 + nc]      = f2bf(acc0[r]);
    dst[(long)(mb + r) * SD + n0 + 16 + nc] = f2bf(acc1[r]);
  }
}

// ==================== kernel 3: S = wq . wk^T ====================
// S[h,b,q,k] f32.  grid (L/64, L/64, H*B).  Kd = 64 (2 wmma steps)
__global__ __launch_bounds__(256) void k_dot(const unsigned short* __restrict__ wq,
                                             const unsigned short* __restrict__ wk,
                                             float* __restrict__ S) {
  __shared__ __align__(16) unsigned short At[64 * 40];
  __shared__ __align__(16) unsigned short Bt[64 * 40];
  int tid = threadIdx.x, lane = tid & 31, w = tid >> 5;
  int l0 = blockIdx.x * 64, n0t = blockIdx.y * 64;
  long z = blockIdx.z;
  const unsigned short* Ab = wq + (z * SL + l0) * SK;
  const unsigned short* Bb = wk + (z * SL + n0t) * SK;
  int msub = (w & 3) * 16, n0 = (w >> 2) * 32;
  v8f acc0 = vzero(), acc1 = vzero();
  for (int kk = 0; kk < SK; kk += 32) {
    ld_bf16_tile64(At, Ab + kk, SK, tid);
    ld_bf16_tile64(Bt, Bb + kk, SK, tid);
    __syncthreads();
    v16bf a  = load_a_frag(At, 40, msub, 0, lane);
    v16bf b0 = load_b_frag(Bt, 40, n0, lane);
    v16bf b1 = load_b_frag(Bt, 40, n0 + 16, lane);
    acc0 = wmma_bf16(a, b0, acc0);
    acc1 = wmma_bf16(a, b1, acc1);
    __syncthreads();
  }
  float* dst = S + (z * SL + l0) * SL + n0t;
  int mb = msub + ((lane >> 4) << 3), nc = lane & 15;
#pragma unroll
  for (int r = 0; r < 8; ++r) {
    dst[(long)(mb + r) * SL + n0 + nc]      = acc0[r];
    dst[(long)(mb + r) * SL + n0 + 16 + nc] = acc1[r];
  }
}

// ==================== kernel 4: row softmax + column-sum ====================
// In-place on S; colsum[h,b,k] += sum_q P[q,k].  grid (L/32, 1, H*B)
__global__ __launch_bounds__(256) void k_softmax(float* __restrict__ S,
                                                 float* __restrict__ colsum) {
  __shared__ float red[32 * 8];
  __shared__ float colacc[8 * SL];   // 8 privatized slices (row & 7)
  int tid = threadIdx.x;
  long z = blockIdx.z;
  int r0 = blockIdx.x * 32;
  int r = tid >> 3, c8 = tid & 7;
  float* row = S + (z * SL + r0 + r) * SL;
#pragma unroll
  for (int i = 0; i < 8 * SL / 256; ++i) colacc[i * 256 + tid] = 0.f;
  float mx = -3.4e38f;
#pragma unroll
  for (int j = 0; j < 64; ++j) mx = fmaxf(mx, row[c8 * 64 + j]);
  red[r * 8 + c8] = mx;
  __syncthreads();
  float rmx = red[r * 8];
#pragma unroll
  for (int j = 1; j < 8; ++j) rmx = fmaxf(rmx, red[r * 8 + j]);
  float ev[64];
  float s = 0.f;
#pragma unroll
  for (int j = 0; j < 64; ++j) {
    float e = __expf(row[c8 * 64 + j] - rmx);
    ev[j] = e; s += e;
  }
  __syncthreads();
  red[r * 8 + c8] = s;
  __syncthreads();
  float rs = 0.f;
#pragma unroll
  for (int j = 0; j < 8; ++j) rs += red[r * 8 + j];
  float inv = 1.0f / rs;
  float* cslice = colacc + (r & 7) * SL;
#pragma unroll
  for (int j = 0; j < 64; ++j) {
    float p = ev[j] * inv;
    row[c8 * 64 + j] = p;
    atomicAdd(&cslice[c8 * 64 + j], p);
  }
  __syncthreads();
  float c0 = 0.f, c1 = 0.f;
#pragma unroll
  for (int sIdx = 0; sIdx < 8; ++sIdx) {
    c0 += colacc[sIdx * SL + tid];
    c1 += colacc[sIdx * SL + tid + 256];
  }
  atomicAdd(&colsum[z * SL + tid], c0);
  atomicAdd(&colsum[z * SL + tid + 256], c1);
}

// ==================== kernel 5: U = query - (renorm(P)) . wv ====================
// grid (L/64, D/64, H*B)
__global__ __launch_bounds__(256) void k_av_sub(const float* __restrict__ P,
                                               const float* __restrict__ colsum,
                                               const unsigned short* __restrict__ wv,
                                               const float* __restrict__ Q,
                                               unsigned short* __restrict__ U) {
  __shared__ __align__(16) unsigned short At[2][64 * 40];
  __shared__ __align__(16) unsigned short Bt[2][64 * 40];
  __shared__ float rcs[SL];
  int tid = threadIdx.x, lane = tid & 31, w = tid >> 5;
  int l0 = blockIdx.x * 64, n0t = blockIdx.y * 64;
  long z = blockIdx.z;
  int b = (int)(z & (SB - 1));
  rcs[tid]       = 1.0f / (1e-9f + colsum[z * SL + tid]);
  rcs[tid + 256] = 1.0f / (1e-9f + colsum[z * SL + tid + 256]);
  __syncthreads();
  const float* Ab = P + (z * SL + l0) * SL;
  const unsigned short* Bv = wv + z * SL * SD + n0t;
  int msub = (w & 3) * 16, n0 = (w >> 2) * 32;
  v8f acc0 = vzero(), acc1 = vzero();
  ld_f32_tile64_scaled(At[0], Ab, SL, rcs, tid);
  ld_bf16_tileT(Bt[0], Bv, SD, tid);
  __syncthreads();
  const int NIT = SL / 32;
  for (int i = 0; i < NIT; ++i) {
    int cur = i & 1;
    if (i + 1 < NIT) {
      int kk = (i + 1) * 32;
      ld_f32_tile64_scaled(At[cur ^ 1], Ab + kk, SL, rcs + kk, tid);
      ld_bf16_tileT(Bt[cur ^ 1], Bv + (long)kk * SD, SD, tid);
    }
    v16bf a  = load_a_frag(At[cur], 40, msub, 0, lane);
    v16bf b0 = load_b_frag(Bt[cur], 40, n0, lane);
    v16bf b1 = load_b_frag(Bt[cur], 40, n0 + 16, lane);
    acc0 = wmma_bf16(a, b0, acc0);
    acc1 = wmma_bf16(a, b1, acc1);
    __syncthreads();
  }
  unsigned short* dst = U + (z * SL + l0) * SD + n0t;
  int mb = msub + ((lane >> 4) << 3), nc = lane & 15;
#pragma unroll
  for (int r = 0; r < 8; ++r) {
    int m = mb + r;
    const float* qrow = Q + ((long)(l0 + m) * SB + b) * SD + n0t;
    dst[(long)m * SD + n0 + nc]      = f2bf(qrow[n0 + nc]      - acc0[r]);
    dst[(long)m * SD + n0 + 16 + nc] = f2bf(qrow[n0 + 16 + nc] - acc1[r]);
  }
}

// ==================== kernel 6: o = relu(U . Wt^T) ====================
// grid (L/64, D/64, H*B)
__global__ __launch_bounds__(256) void k_trans_relu(const unsigned short* __restrict__ U,
                                                    const float* __restrict__ Wt,
                                                    unsigned short* __restrict__ O) {
  __shared__ __align__(16) unsigned short At[2][64 * 40];
  __shared__ __align__(16) unsigned short Bt[2][64 * 40];
  int tid = threadIdx.x, lane = tid & 31, w = tid >> 5;
  int l0 = blockIdx.x * 64, n0t = blockIdx.y * 64;
  long z = blockIdx.z;
  int h = (int)(z >> 5);
  const unsigned short* Ab = U + (z * SL + l0) * SD;
  const float* Wb = Wt + (long)h * SD * SD + (long)n0t * SD;
  int msub = (w & 3) * 16, n0 = (w >> 2) * 32;
  v8f acc0 = vzero(), acc1 = vzero();
  ld_bf16_tile64(At[0], Ab, SD, tid);
  ld_f32_tile64(Bt[0], Wb, SD, tid);
  __syncthreads();
  const int NIT = SD / 32;
  for (int i = 0; i < NIT; ++i) {
    int cur = i & 1;
    if (i + 1 < NIT) {
      ld_bf16_tile64(At[cur ^ 1], Ab + (i + 1) * 32, SD, tid);
      ld_f32_tile64(Bt[cur ^ 1], Wb + (i + 1) * 32, SD, tid);
    }
    v16bf a  = load_a_frag(At[cur], 40, msub, 0, lane);
    v16bf b0 = load_b_frag(Bt[cur], 40, n0, lane);
    v16bf b1 = load_b_frag(Bt[cur], 40, n0 + 16, lane);
    acc0 = wmma_bf16(a, b0, acc0);
    acc1 = wmma_bf16(a, b1, acc1);
    __syncthreads();
  }
  unsigned short* dst = O + (z * SL + l0) * SD + n0t;
  int mb = msub + ((lane >> 4) << 3), nc = lane & 15;
#pragma unroll
  for (int r = 0; r < 8; ++r) {
    dst[(long)(mb + r) * SD + n0 + nc]      = f2bf(fmaxf(acc0[r], 0.f));
    dst[(long)(mb + r) * SD + n0 + 16 + nc] = f2bf(fmaxf(acc1[r], 0.f));
  }
}

// ==================== kernel 7: fused FFN + LayerNorm + head concat ====================
// 32 rows per workgroup.  grid (L/32, 1, H*B)
__global__ __launch_bounds__(256) void k_ffn_ln(const unsigned short* __restrict__ O,
                                                const float* __restrict__ W1,
                                                const float* __restrict__ b1,
                                                const float* __restrict__ W2,
                                                const float* __restrict__ b2,
                                                const float* __restrict__ lng,
                                                const float* __restrict__ lnb,
                                                float* __restrict__ out) {
  __shared__ __align__(16) char smem[55296];
  // step1 double buffers (alias region [0, 40960) also used by Bt2 / reductions)
  unsigned short* At0  = (unsigned short*)(smem);                 // 32x40
  unsigned short* At1  = (unsigned short*)(smem + 2560);
  unsigned short* Bt10 = (unsigned short*)(smem + 8192);          // 64x40
  unsigned short* Bt11 = (unsigned short*)(smem + 8192 + 5120);
  unsigned short* Bt2  = (unsigned short*)(smem);                 // 512x40 (step2)
  float* redS = (float*)(smem);                                   // 32x64 (epilogue)
  float* redQ = (float*)(smem + 8192);                            // 32x64
  unsigned short* hff = (unsigned short*)(smem + 49152);          // 32x72
  float* stat = (float*)(smem + 53760);                           // mean[32], rstd[32]

  int tid = threadIdx.x, lane = tid & 31, w = tid >> 5;
  int l0 = blockIdx.x * 32;
  long z = blockIdx.z;
  int b = (int)(z & (SB - 1)), h = (int)(z >> 5);
  const unsigned short* Ob = O + (z * SL + l0) * SD;
  const float* W1g = W1 + (long)h * SF * SD;
  const float* b1g = b1 + (long)h * SF;
  const float* W2g = W2 + (long)h * SD * SF;
  const float* b2g = b2 + (long)h * SD;
  const float* gg  = lng + (long)h * SD;
  const float* bb  = lnb + (long)h * SD;

  // ---- step 1: hff = relu(o . W1^T + b1)  (32 x 64), double-buffered ----
  int msub = (w & 1) * 16, n0 = (w >> 1) * 16;
  v8f acc = vzero();
  unsigned short* Ax[2]  = { At0, At1 };
  unsigned short* B1x[2] = { Bt10, Bt11 };
  ld_bf16_tile32(Ax[0], Ob, SD, tid);
  ld_f32_tile64(B1x[0], W1g, SD, tid);
  __syncthreads();
  const int NIT = SD / 32;
  for (int i = 0; i < NIT; ++i) {
    int cur = i & 1;
    if (i + 1 < NIT) {
      ld_bf16_tile32(Ax[cur ^ 1], Ob + (i + 1) * 32, SD, tid);
      ld_f32_tile64(B1x[cur ^ 1], W1g + (i + 1) * 32, SD, tid);
    }
    v16bf a  = load_a_frag(Ax[cur], 40, msub, 0, lane);
    v16bf bf = load_b_frag(B1x[cur], 40, n0, lane);
    acc = wmma_bf16(a, bf, acc);
    __syncthreads();
  }
  {
    int mb = msub + ((lane >> 4) << 3), nc = lane & 15;
#pragma unroll
    for (int r = 0; r < 8; ++r) {
      int n = n0 + nc;
      hff[(mb + r) * 72 + n] = f2bf(fmaxf(acc[r] + b1g[n], 0.f));
    }
  }
  __syncthreads();

  // ---- step 2: s2 = hff . W2^T (+b2 in epilogue), all 512 cols in registers ----
  int nBase = (w >> 1) * 128;
  v8f acc2[8];
#pragma unroll
  for (int j = 0; j < 8; ++j) acc2[j] = vzero();
  for (int kk = 0; kk < SF; kk += 32) {
    {  // load 512x32 slab of W2 into Bt2 (bf16)
      int rbase = tid >> 2, seg = (tid & 3) * 8;
#pragma unroll
      for (int rep = 0; rep < 8; ++rep) {
        int n = rbase + rep * 64;
        const float* g = W2g + (long)n * SF + kk + seg;
        float4 f0 = *(const float4*)(g);
        float4 f1 = *(const float4*)(g + 4);
        *(uint4*)(Bt2 + n * 40 + seg) = pack8(f0.x,f0.y,f0.z,f0.w,f1.x,f1.y,f1.z,f1.w);
      }
    }
    __syncthreads();
    v16bf a = load_a_frag(hff, 72, msub, kk, lane);
#pragma unroll
    for (int j = 0; j < 8; ++j) {
      v16bf bf = load_b_frag(Bt2, 40, nBase + j * 16, lane);
      acc2[j] = wmma_bf16(a, bf, acc2[j]);
    }
    __syncthreads();
  }

  // ---- epilogue: x = o + s2 + b2; LayerNorm over 512; concat heads ----
  int mb = msub + ((lane >> 4) << 3), nc = lane & 15;
  int wcol = (w >> 1) * 16 + nc;
#pragma unroll
  for (int r = 0; r < 8; ++r) {
    int row = mb + r;
    const unsigned short* orow = O + (z * SL + l0 + row) * SD;
    float sm = 0.f, sq = 0.f;
#pragma unroll
    for (int j = 0; j < 8; ++j) {
      int n = nBase + j * 16 + nc;
      float v = acc2[j][r] + b2g[n] + bf2f(orow[n]);
      sm += v; sq += v * v;
    }
    redS[row * 64 + wcol] = sm;
    redQ[row * 64 + wcol] = sq;
  }
  __syncthreads();
  if (tid < 32) {
    float sm = 0.f, sq = 0.f;
    for (int c = 0; c < 64; ++c) { sm += redS[tid * 64 + c]; sq += redQ[tid * 64 + c]; }
    float mean = sm * (1.0f / SD);
    float var  = sq * (1.0f / SD) - mean * mean;
    stat[tid]      = mean;
    stat[32 + tid] = rsqrtf(fmaxf(var, 0.f) + 1e-5f);
  }
  __syncthreads();
#pragma unroll
  for (int r = 0; r < 8; ++r) {
    int row = mb + r;
    const unsigned short* orow = O + (z * SL + l0 + row) * SD;
    float mean = stat[row], rstd = stat[32 + row];
    float* orow_out = out + ((long)(l0 + row) * SB + b) * (SH * SD) + (long)h * SD;
#pragma unroll
    for (int j = 0; j < 8; ++j) {
      int n = nBase + j * 16 + nc;
      float v = acc2[j][r] + b2g[n] + bf2f(orow[n]);
      orow_out[n] = (v - mean) * rstd * gg[n] + bb[n];
    }
  }
}

// ==================== host launcher ====================
extern "C" void kernel_launch(void* const* d_in, const int* in_sizes, int n_in,
                              void* d_out, int out_size, void* d_ws, size_t ws_size,
                              hipStream_t stream) {
  (void)in_sizes; (void)n_in; (void)out_size; (void)ws_size;
  const float* query = (const float*)d_in[0];
  const float* key   = (const float*)d_in[1];
  const float* value = (const float*)d_in[2];
  const float* WK    = (const float*)d_in[3];
  const float* WQ    = (const float*)d_in[4];
  const float* WV    = (const float*)d_in[5];
  const float* Wt    = (const float*)d_in[6];
  const float* W1    = (const float*)d_in[7];
  const float* b1    = (const float*)d_in[8];
  const float* W2    = (const float*)d_in[9];
  const float* b2    = (const float*)d_in[10];
  const float* lng   = (const float*)d_in[11];
  const float* lnb   = (const float*)d_in[12];
  float* out = (float*)d_out;
  char* ws = (char*)d_ws;

  // workspace layout (bytes); o aliases the dead score buffer S
  constexpr size_t SZ_QK = (size_t)SH * SB * SL * SK * 2;  //  16 MiB each (bf16)
  constexpr size_t SZ_WV = (size_t)SH * SB * SL * SD * 2;  // 128 MiB (bf16)
  constexpr size_t SZ_S  = (size_t)SH * SB * SL * SL * 4;  // 256 MiB (f32)
  unsigned short* wqn = (unsigned short*)(ws);
  unsigned short* wkn = (unsigned short*)(ws + SZ_QK);
  unsigned short* wv  = (unsigned short*)(ws + 2 * SZ_QK);
  float*          S   = (float*)(ws + 2 * SZ_QK + SZ_WV);
  unsigned short* U   = (unsigned short*)(ws + 2 * SZ_QK + SZ_WV + SZ_S);
  float*       colsum = (float*)(ws + 2 * SZ_QK + 2 * SZ_WV + SZ_S);
  unsigned short* Obuf = (unsigned short*)S;  // alias: S dead after k_av_sub

  dim3 blk(256);
  k_proj_norm<<<dim3(SL / 64, SB, SH), blk, 0, stream>>>(query, WQ, wqn);
  k_proj_norm<<<dim3(SL / 64, SB, SH), blk, 0, stream>>>(key,   WK, wkn);
  k_proj_v   <<<dim3(SL / 64, SD / 64, SH * SB), blk, 0, stream>>>(value, WV, wv);
  k_dot      <<<dim3(SL / 64, SL / 64, SH * SB), blk, 0, stream>>>(wqn, wkn, S);
  k_zero     <<<dim3((SH * SB * SL + 255) / 256), blk, 0, stream>>>(colsum, SH * SB * SL);
  k_softmax  <<<dim3(SL / 32, 1, SH * SB), blk, 0, stream>>>(S, colsum);
  k_av_sub   <<<dim3(SL / 64, SD / 64, SH * SB), blk, 0, stream>>>(S, colsum, wv, query, U);
  k_trans_relu<<<dim3(SL / 64, SD / 64, SH * SB), blk, 0, stream>>>(U, Wt, Obuf);
  k_ffn_ln   <<<dim3(SL / 32, 1, SH * SB), blk, 0, stream>>>(Obuf, W1, b1, W2, b2, lng, lnb, out);
}
